// HAB_24833500905603
// MI455X (gfx1250) — compile-verified, matrix operand order checked by reading, not run on hardware
//
#include <hip/hip_runtime.h>
#include <math.h>

// ---------------------------------------------------------------------------
// Problem constants (from reference)
// ---------------------------------------------------------------------------
#define BB     8
#define HH     96
#define WW     96
#define NPIX   (HH*WW)          // 9216
#define DIM    180
#define MROWS  (BB*NPIX)        // 73728 tokens
#define CR     60
#define HID    360
#define HID2   720
#define PDIM   16
#define KS     13
#define WF     (WW/2+1)         // 49 rfft bins
#define KP180  192              // K=180 padded to 32
#define KP360  384              // K=360 padded to 32
#define FFT_RES 0.15f
#define CONV_SCALE 0.01f

// ---------------------------------------------------------------------------
// Workspace layout (bytes). Total ~326 MB; regions reused across stages.
// ---------------------------------------------------------------------------
static const size_t SZ_IMG  = (size_t)MROWS * DIM * 4;          // 53,084,160
static const size_t SZ_FW   = (size_t)BB*HH*WF*DIM * 4;         // 27,095,040
static const size_t OFF_XN   = 0;                               // xn (LN1 out, channel-last image)
static const size_t OFF_FWR  = OFF_XN  + SZ_IMG;                // Xw real | later conv_x (y2)
static const size_t OFF_FWI  = OFF_FWR + SZ_FW;                 // Xw imag | (conv_x tail)
static const size_t OFF_F2R  = OFF_FWI + SZ_FW;                 // X2 real | later y1 (CAB mid)
static const size_t OFF_F2I  = OFF_F2R + SZ_FW;                 // X2 imag | later lk+dyn
static const size_t OFF_BAND = OFF_F2I + SZ_FW;                 // band f32 | later xm
static const size_t OFF_XFFT = OFF_BAND+ SZ_IMG;                // x_fft accumulator
static const size_t OFF_A16  = OFF_XFFT+ SZ_IMG;                // f16 activations [M,384] max
static const size_t OFF_W16  = OFF_A16 + (size_t)MROWS*KP360*2; // f16 padded weights
static const size_t OFF_SML  = OFF_W16 + 1048576;               // tables + tiny tensors
static const size_t OFF_T16  = 0;  // fc1 output f16 [M,720] overlays freed xn/fw/f2 regions

// small-region float offsets
#define SM_CW   0        // 49*96 cos (w-DFT)
#define SM_SW   4704
#define SM_CH   9408     // 96*96 cos (h-DFT)
#define SM_SH   18624
#define SM_IMGM 27840    // [B,180] mean(x_img)
#define SM_GATE 29280    // [B,3]
#define SM_GAP  29312    // [B,180]
#define SM_SIG  30752    // [B,180]
#define SM_X1M  32192    // [B,16]
#define SM_DK   32320    // [B,144] dynamic 3x3 kernels

// w16-region half offsets (fc1 padded to 736 rows so 32-wide N tiles stay in-bounds)
#define W16_BAND 0          // 3 x 192*192
#define W16_AGGR 110592     // 192*192
#define W16_FC1  147456     // 736*192
#define W16_FC2  288768     // 192*384
#define FC1_NPAD 736

typedef _Float16 h8   __attribute__((ext_vector_type(8)));
typedef _Float16 v16h __attribute__((ext_vector_type(16)));
typedef float    v8f  __attribute__((ext_vector_type(8)));

__device__ __forceinline__ float gelu_f(float v) {
  return 0.5f * v * (1.f + erff(v * 0.70710678118654752f));
}
__device__ __forceinline__ float sigm_f(float v) { return 1.f / (1.f + expf(-v)); }

// ---------------------------------------------------------------------------
// Trig tables for 96-point DFTs
// ---------------------------------------------------------------------------
__global__ void tables_k(float* __restrict__ sml) {
  int gid = blockIdx.x * 256 + threadIdx.x;
  const float TW = 6.283185307179586f / 96.f;
  if (gid < 4704) {
    int v = gid / 96, w = gid % 96; float t = TW * (float)(v * w);
    sml[SM_CW + gid] = cosf(t); sml[SM_SW + gid] = sinf(t);
  } else if (gid < 4704 + 9216) {
    int i = gid - 4704; int u = i / 96, h = i % 96; float t = TW * (float)(u * h);
    sml[SM_CH + i] = cosf(t); sml[SM_SH + i] = sinf(t);
  }
}

// f32 [N,K] weight -> f16 [Npad,Kp] zero-padded (rows = output features)
__global__ void wconv_k(const float* __restrict__ W, _Float16* __restrict__ out,
                        int N, int K, int Kp, int total) {
  int gid = blockIdx.x * 256 + threadIdx.x;
  if (gid >= total) return;
  int n = gid / Kp, kk = gid % Kp;
  out[gid] = (n < N && kk < K) ? (_Float16)W[n * K + kk] : (_Float16)0.f;
}

// ---------------------------------------------------------------------------
// LayerNorm: one wave (32 lanes) per token row. Writes xn and x_fft init.
// ---------------------------------------------------------------------------
__global__ __launch_bounds__(256) void ln1_k(const float* __restrict__ x,
                                             const float* __restrict__ g,
                                             const float* __restrict__ b,
                                             float* __restrict__ xn,
                                             float* __restrict__ xfft) {
  int row = blockIdx.x * 8 + (threadIdx.x >> 5);
  int lane = threadIdx.x & 31;
  const float* xr = x + (size_t)row * DIM;
  float s = 0.f, s2 = 0.f;
  for (int i = lane; i < DIM; i += 32) { float v = xr[i]; s += v; s2 += v * v; }
  for (int o = 16; o; o >>= 1) { s += __shfl_xor(s, o, 32); s2 += __shfl_xor(s2, o, 32); }
  float mean = s * (1.f / DIM);
  float inv  = rsqrtf(s2 * (1.f / DIM) - mean * mean + 1e-5f);
  for (int i = lane; i < DIM; i += 32) {
    float v = (xr[i] - mean) * inv * g[i] + b[i];
    xn[(size_t)row * DIM + i] = v;
    xfft[(size_t)row * DIM + i] = v;
  }
}

// LN2: reads xm, writes padded f16 [M,192]
__global__ __launch_bounds__(256) void ln2_k(const float* __restrict__ xm,
                                             const float* __restrict__ g,
                                             const float* __restrict__ b,
                                             _Float16* __restrict__ a16) {
  int row = blockIdx.x * 8 + (threadIdx.x >> 5);
  int lane = threadIdx.x & 31;
  const float* xr = xm + (size_t)row * DIM;
  float s = 0.f, s2 = 0.f;
  for (int i = lane; i < DIM; i += 32) { float v = xr[i]; s += v; s2 += v * v; }
  for (int o = 16; o; o >>= 1) { s += __shfl_xor(s, o, 32); s2 += __shfl_xor(s2, o, 32); }
  float mean = s * (1.f / DIM);
  float inv  = rsqrtf(s2 * (1.f / DIM) - mean * mean + 1e-5f);
  for (int i = lane; i < KP180; i += 32) {
    float v = (i < DIM) ? (xr[i] - mean) * inv * g[i] + b[i] : 0.f;
    a16[(size_t)row * KP180 + i] = (_Float16)v;
  }
}

// mean over the 9216 pixels for channels [0,nc) of a channel-last image
__global__ __launch_bounds__(256) void meanpix_k(const float* __restrict__ in,
                                                 float* __restrict__ out,
                                                 int nc, int stride) {
  int b = blockIdx.x / nc, c = blockIdx.x % nc;
  __shared__ float sm[256];
  float s = 0.f;
  for (int n = threadIdx.x; n < NPIX; n += 256)
    s += in[((size_t)b * NPIX + n) * stride + c];
  sm[threadIdx.x] = s; __syncthreads();
  for (int st = 128; st; st >>= 1) {
    if (threadIdx.x < st) sm[threadIdx.x] += sm[threadIdx.x + st];
    __syncthreads();
  }
  if (!threadIdx.x) out[blockIdx.x] = sm[0] * (1.f / NPIX);
}

// gates = softmax_k(sigmoid(imgmean @ gate_w.T + gate_b))
__global__ void gates_k(const float* __restrict__ sml, const float* __restrict__ gw,
                        const float* __restrict__ gb, float* __restrict__ smlw) {
  int b = threadIdx.x; if (b >= BB) return;
  float sg[3];
  for (int k = 0; k < 3; ++k) {
    float a = gb[k];
    for (int c = 0; c < DIM; ++c) a += sml[SM_IMGM + b * DIM + c] * gw[k * DIM + c];
    sg[k] = sigm_f(a);
  }
  float mx = fmaxf(sg[0], fmaxf(sg[1], sg[2]));
  float e0 = expf(sg[0]-mx), e1 = expf(sg[1]-mx), e2 = expf(sg[2]-mx), se = e0+e1+e2;
  smlw[SM_GATE + b*3+0] = e0/se; smlw[SM_GATE + b*3+1] = e1/se; smlw[SM_GATE + b*3+2] = e2/se;
}

// ---------------------------------------------------------------------------
// Separable DFT stages (unnormalized fw; 1/(H*W) folded into inverse-w)
// ---------------------------------------------------------------------------
__global__ __launch_bounds__(256) void dftw_k(const float* __restrict__ xn,
                                              const float* __restrict__ sml,
                                              float* __restrict__ XR, float* __restrict__ XI) {
  int gid = blockIdx.x * 256 + threadIdx.x;
  int c = gid % DIM; int t = gid / DIM;
  int v = t % WF; t /= WF; int h = t % HH; int b = t / HH;
  float sR = 0.f, sI = 0.f;
  const float* xb = xn + ((size_t)(b * HH + h) * WW) * DIM + c;
  for (int w = 0; w < WW; ++w) {
    float xv = xb[(size_t)w * DIM];
    sR += xv * sml[SM_CW + v * 96 + w];
    sI -= xv * sml[SM_SW + v * 96 + w];
  }
  XR[gid] = sR; XI[gid] = sI;
}

__global__ __launch_bounds__(256) void dfth_k(const float* __restrict__ XR,
                                              const float* __restrict__ XI,
                                              const float* __restrict__ sml,
                                              float* __restrict__ YR, float* __restrict__ YI) {
  int gid = blockIdx.x * 256 + threadIdx.x;
  int c = gid % DIM; int t = gid / DIM;
  int v = t % WF; t /= WF; int u = t % HH; int b = t / HH;
  float sR = 0.f, sI = 0.f;
  for (int h = 0; h < HH; ++h) {
    size_t idx = ((size_t)(b * HH + h) * WF + v) * DIM + c;
    float R = XR[idx], I = XI[idx];
    float ch = sml[SM_CH + u * 96 + h], sh = sml[SM_SH + u * 96 + h];
    sR += ch * R + sh * I;   // e^{-i}
    sI += ch * I - sh * R;
  }
  YR[gid] = sR; YI[gid] = sI;
}

// inverse over h with radial band mask applied (mask from u,v grid)
__global__ __launch_bounds__(256) void invh_k(const float* __restrict__ XR,
                                              const float* __restrict__ XI,
                                              const float* __restrict__ sml,
                                              float* __restrict__ YR, float* __restrict__ YI,
                                              int band) {
  int gid = blockIdx.x * 256 + threadIdx.x;
  int c = gid % DIM; int t = gid / DIM;
  int v = t % WF; t /= WF; int h = t % HH; int b = t / HH;
  float xx = (float)v / 48.f;
  float sR = 0.f, sI = 0.f;
  for (int u = 0; u < HH; ++u) {
    float yy = -1.f + 2.f * (float)u / 95.f;
    float rr = sqrtf(yy * yy + xx * xx);
    bool m = (band == 0) ? (rr <= 0.25f)
           : (band == 1) ? (rr > 0.25f && rr <= 0.6f)
                         : (rr > 0.6f);
    if (!m) continue;
    size_t idx = ((size_t)(b * HH + u) * WF + v) * DIM + c;
    float R = XR[idx], I = XI[idx];
    float ch = sml[SM_CH + u * 96 + h], sh = sml[SM_SH + u * 96 + h];
    sR += ch * R - sh * I;   // e^{+i}
    sI += ch * I + sh * R;
  }
  YR[gid] = sR; YI[gid] = sI;
}

// inverse rfft over w (hermitian doubling, ortho 1/(H*W))
__global__ __launch_bounds__(256) void invw_k(const float* __restrict__ YR,
                                              const float* __restrict__ YI,
                                              const float* __restrict__ sml,
                                              float* __restrict__ band) {
  int gid = blockIdx.x * 256 + threadIdx.x;
  int c = gid % DIM; int t = gid / DIM;
  int w = t % WW; t /= WW; int h = t % HH; int b = t / HH;
  float s = 0.f;
  for (int v = 0; v < WF; ++v) {
    size_t idx = ((size_t)(b * HH + h) * WF + v) * DIM + c;
    float dup = (v == 0 || v == 48) ? 1.f : 2.f;
    s += dup * (YR[idx] * sml[SM_CW + v * 96 + w] - YI[idx] * sml[SM_SW + v * 96 + w]);
  }
  band[gid] = s * (1.f / (float)(HH * WW));
}

// f32 [M,180] -> padded f16 [M,192]
__global__ __launch_bounds__(256) void cvt16_k(const float* __restrict__ in,
                                               _Float16* __restrict__ out) {
  int gid = blockIdx.x * 256 + threadIdx.x;
  int n = gid % KP180, m = gid / KP180;
  out[gid] = (n < DIM) ? (_Float16)in[(size_t)m * DIM + n] : (_Float16)0.f;
}

// ya16 = concat([lk+dyn (16ch), x2 = xn[16:180]]) padded to 192
__global__ __launch_bounds__(256) void ya16_k(const float* __restrict__ lk,
                                              const float* __restrict__ xn,
                                              _Float16* __restrict__ out) {
  int gid = blockIdx.x * 256 + threadIdx.x;
  int n = gid % KP180, m = gid / KP180;
  float v = 0.f;
  if (n < PDIM)      v = lk[(size_t)m * PDIM + n];
  else if (n < DIM)  v = xn[(size_t)m * DIM + n];
  out[gid] = (_Float16)v;
}

// ---------------------------------------------------------------------------
// WMMA GEMM: out[M,N] = A[M,Kp] @ Bt[Npad,Kp]^T.
// One 32x32 output tile per wave32 (2x2 WMMA tiles): 8 b128 loads feed
// 4 v_wmma_f32_16x16x32_f16 per K-chunk (2x register operand reuse).
// mode 0: xfft += gate*FFT_RES*(acc+bias)                 (band mixing)
// mode 1: xm    = shortcut + (acc+bias) + 0.01*conv_x     (aggregation)
// mode 2: t16   = (half)(acc+bias)                        (fc1)
// mode 3: d_out = xm + acc + bias                         (fc2 + residual)
// ---------------------------------------------------------------------------
__global__ __launch_bounds__(256) void gemm_wmma_k(
    const _Float16* __restrict__ A, const _Float16* __restrict__ Bt,
    int Kp, int Nvalid, int mode,
    const float* __restrict__ bias,
    const float* __restrict__ add0, const float* __restrict__ add1,
    float* __restrict__ outF, _Float16* __restrict__ outH,
    const float* __restrict__ gates, int bandk, int outStride) {
  const int lane = threadIdx.x & 31;
  const int mt = blockIdx.x * 8 + (threadIdx.x >> 5);
  const int m0 = mt * 32, n0 = blockIdx.y * 32;
  const int kh = (lane >> 4) * 8;               // K sub-offset per half-wave
  const _Float16* Ar0 = A  + (size_t)(m0 + (lane & 15)) * Kp + kh;
  const _Float16* Ar1 = Ar0 + (size_t)16 * Kp;
  const _Float16* Br0 = Bt + (size_t)(n0 + (lane & 15)) * Kp + kh;
  const _Float16* Br1 = Br0 + (size_t)16 * Kp;
  v8f c00 = {}, c01 = {}, c10 = {}, c11 = {};
  for (int k0 = 0; k0 < Kp; k0 += 32) {
    __builtin_prefetch(Ar0 + k0 + 32, 0, 1);    // global_prefetch next A chunk
    union { v16h v; h8 h[2]; } a0, a1, b0, b1;
    a0.h[0] = *(const h8*)(Ar0 + k0);  a0.h[1] = *(const h8*)(Ar0 + k0 + 16);
    a1.h[0] = *(const h8*)(Ar1 + k0);  a1.h[1] = *(const h8*)(Ar1 + k0 + 16);
    b0.h[0] = *(const h8*)(Br0 + k0);  b0.h[1] = *(const h8*)(Br0 + k0 + 16);
    b1.h[0] = *(const h8*)(Br1 + k0);  b1.h[1] = *(const h8*)(Br1 + k0 + 16);
    c00 = __builtin_amdgcn_wmma_f32_16x16x32_f16(false, a0.v, false, b0.v, (short)0, c00, false, false);
    c01 = __builtin_amdgcn_wmma_f32_16x16x32_f16(false, a0.v, false, b1.v, (short)0, c01, false, false);
    c10 = __builtin_amdgcn_wmma_f32_16x16x32_f16(false, a1.v, false, b0.v, (short)0, c10, false, false);
    c11 = __builtin_amdgcn_wmma_f32_16x16x32_f16(false, a1.v, false, b1.v, (short)0, c11, false, false);
  }
  v8f cc[2][2] = {{c00, c01}, {c10, c11}};
#pragma unroll
  for (int j = 0; j < 2; ++j) {
    const int nn = n0 + j * 16 + (lane & 15);
    if (nn >= Nvalid) continue;
#pragma unroll
    for (int i = 0; i < 2; ++i) {
      const int mbase = m0 + i * 16 + ((lane >> 4) << 3);  // lanes 16-31: rows +8
#pragma unroll
      for (int r = 0; r < 8; ++r) {
        int m = mbase + r;
        float acc = cc[i][j][r] + bias[nn];
        size_t o = (size_t)m * DIM + nn;
        if (mode == 0) {
          outF[o] += gates[(m / NPIX) * 3 + bandk] * FFT_RES * acc;
        } else if (mode == 1) {
          outF[o] = add0[o] + acc + CONV_SCALE * add1[o];
        } else if (mode == 2) {
          outH[(size_t)m * outStride + nn] = (_Float16)acc;
        } else {
          outF[o] = add0[o] + acc;
        }
      }
    }
  }
}

// ---------------------------------------------------------------------------
// CAB branch (channel-last images)
// ---------------------------------------------------------------------------
__global__ __launch_bounds__(256) void cab1_k(const float* __restrict__ xfft,
                                              const float* __restrict__ w1,
                                              const float* __restrict__ b1,
                                              float* __restrict__ y1) {
  int gid = blockIdx.x * 256 + threadIdx.x;
  int o = gid % CR; int t = gid / CR;
  int w = t % WW; t /= WW; int h = t % HH; int b = t / HH;
  float acc = b1[o];
  for (int dy = 0; dy < 3; ++dy) {
    int y = h + dy - 1; if (y < 0 || y >= HH) continue;
    for (int dx = 0; dx < 3; ++dx) {
      int x = w + dx - 1; if (x < 0 || x >= WW) continue;
      const float* xp = xfft + ((size_t)(b * HH + y) * WW + x) * DIM;
      const float* wp = w1 + (size_t)o * DIM * 9 + dy * 3 + dx;
      for (int cc = 0; cc < DIM; ++cc) acc += xp[cc] * wp[(size_t)cc * 9];
    }
  }
  y1[gid] = gelu_f(acc);
}

__global__ __launch_bounds__(256) void cab2_k(const float* __restrict__ y1,
                                              const float* __restrict__ w2,
                                              const float* __restrict__ b2,
                                              float* __restrict__ y2) {
  int gid = blockIdx.x * 256 + threadIdx.x;
  int o = gid % DIM; int t = gid / DIM;
  int w = t % WW; t /= WW; int h = t % HH; int b = t / HH;
  float acc = b2[o];
  for (int dy = 0; dy < 3; ++dy) {
    int y = h + dy - 1; if (y < 0 || y >= HH) continue;
    for (int dx = 0; dx < 3; ++dx) {
      int x = w + dx - 1; if (x < 0 || x >= WW) continue;
      const float* yp = y1 + ((size_t)(b * HH + y) * WW + x) * CR;
      const float* wp = w2 + (size_t)o * CR * 9 + dy * 3 + dx;
      for (int cc = 0; cc < CR; ++cc) acc += yp[cc] * wp[(size_t)cc * 9];
    }
  }
  y2[gid] = acc;
}

// ECA: 1D conv over channel axis of gap, sigmoid
__global__ void eca_k(const float* __restrict__ sml, const float* __restrict__ ew,
                      float* __restrict__ smlw) {
  for (int idx = threadIdx.x; idx < BB * DIM; idx += 256) {
    int b = idx / DIM, o = idx % DIM;
    float z = 0.f;
    for (int i = 0; i < 5; ++i) {
      int p = o - 2 + i;
      if (p >= 0 && p < DIM) z += sml[SM_GAP + b * DIM + p] * ew[i];
    }
    smlw[SM_SIG + idx] = sigm_f(z);
  }
}

__global__ __launch_bounds__(256) void ecascale_k(float* __restrict__ y2,
                                                  const float* __restrict__ sml) {
  int gid = blockIdx.x * 256 + threadIdx.x;
  int c = gid % DIM;
  int b = gid / (NPIX * DIM);
  y2[gid] *= sml[SM_SIG + b * DIM + c];
}

// ---------------------------------------------------------------------------
// Attention branch: dynamic-kernel MLP + 13x13 partial-LK conv + dyn depthwise
// ---------------------------------------------------------------------------
__global__ void dwcmlp_k(const float* __restrict__ sml,
                         const float* __restrict__ w1, const float* __restrict__ b1,
                         const float* __restrict__ w2, const float* __restrict__ b2,
                         float* __restrict__ smlw) {
  __shared__ float hdn[BB * 8];
  if (threadIdx.x < BB * 8) {
    int b = threadIdx.x >> 3, j = threadIdx.x & 7;
    float a = b1[j];
    for (int p = 0; p < PDIM; ++p) a += sml[SM_X1M + b * PDIM + p] * w1[j * PDIM + p];
    hdn[threadIdx.x] = gelu_f(a);
  }
  __syncthreads();
  for (int idx = threadIdx.x; idx < BB * 144; idx += 256) {
    int b = idx / 144, r = idx % 144;
    float v = b2[r];
    for (int j = 0; j < 8; ++j) v += hdn[b * 8 + j] * w2[r * 8 + j];
    smlw[SM_DK + idx] = v;
  }
}

__global__ __launch_bounds__(256) void lkdyn_k(const float* __restrict__ xn,
                                               const float* __restrict__ plk,
                                               const float* __restrict__ sml,
                                               float* __restrict__ out) {
  int gid = blockIdx.x * 256 + threadIdx.x;
  int p = gid % PDIM; int t = gid / PDIM;
  int w = t % WW; t /= WW; int h = t % HH; int b = t / HH;
  float acc = 0.f;
  // 13x13 conv over the 16 low channels of x_img (no bias)
  for (int cc = 0; cc < PDIM; ++cc) {
    const float* kp = plk + ((size_t)p * PDIM + cc) * (KS * KS);
    for (int dy = 0; dy < KS; ++dy) {
      int y = h + dy - 6; if (y < 0 || y >= HH) continue;
      for (int dx = 0; dx < KS; ++dx) {
        int x = w + dx - 6; if (x < 0 || x >= WW) continue;
        acc += xn[((size_t)(b * HH + y) * WW + x) * DIM + cc] * kp[dy * KS + dx];
      }
    }
  }
  // dynamic 3x3 depthwise
  const float* dk = sml + SM_DK + (b * PDIM + p) * 9;
  for (int dy = 0; dy < 3; ++dy) {
    int y = h + dy - 1; if (y < 0 || y >= HH) continue;
    for (int dx = 0; dx < 3; ++dx) {
      int x = w + dx - 1; if (x < 0 || x >= WW) continue;
      acc += xn[((size_t)(b * HH + y) * WW + x) * DIM + p] * dk[dy * 3 + dx];
    }
  }
  out[gid] = acc;
}

// ---------------------------------------------------------------------------
// FFN tail: depthwise 3x3 over f16 t, SiLU-GLU, write padded f16 [M,384]
// ---------------------------------------------------------------------------
__global__ __launch_bounds__(256) void dwglu_k(const _Float16* __restrict__ t16,
                                               const float* __restrict__ dww,
                                               const float* __restrict__ dwb,
                                               _Float16* __restrict__ pg) {
  int gid = blockIdx.x * 256 + threadIdx.x;
  int j = gid % KP360; int m = gid / KP360;
  if (j >= HID) { pg[gid] = (_Float16)0.f; return; }
  int w = m % WW, h = (m / WW) % HH, b = m / NPIX;
  float val[2];
  for (int s = 0; s < 2; ++s) {
    int ch = j + s * HID;
    float acc = dwb[ch];
    const float* kp = dww + (size_t)ch * 9;
    for (int dy = 0; dy < 3; ++dy) {
      int y = h + dy - 1; if (y < 0 || y >= HH) continue;
      for (int dx = 0; dx < 3; ++dx) {
        int x = w + dx - 1; if (x < 0 || x >= WW) continue;
        acc += (float)t16[((size_t)(b * HH + y) * WW + x) * HID2 + ch] * kp[dy * 3 + dx];
      }
    }
    val[s] = acc;
  }
  float g = val[1];
  pg[gid] = (_Float16)(val[0] * g * sigm_f(g));
}

// ---------------------------------------------------------------------------
// Host-side orchestration
// ---------------------------------------------------------------------------
extern "C" void kernel_launch(void* const* d_in, const int* in_sizes, int n_in,
                              void* d_out, int out_size, void* d_ws, size_t ws_size,
                              hipStream_t stream) {
  (void)in_sizes; (void)n_in; (void)out_size; (void)ws_size;
  const float* x      = (const float*)d_in[0];
  const float* ln1_g  = (const float*)d_in[3];
  const float* ln1_b  = (const float*)d_in[4];
  const float* band_w = (const float*)d_in[5];
  const float* band_b = (const float*)d_in[6];
  const float* gate_w = (const float*)d_in[7];
  const float* gate_b = (const float*)d_in[8];
  const float* cab_w1 = (const float*)d_in[9];
  const float* cab_b1 = (const float*)d_in[10];
  const float* cab_w2 = (const float*)d_in[11];
  const float* cab_b2 = (const float*)d_in[12];
  const float* eca_w  = (const float*)d_in[13];
  const float* plk    = (const float*)d_in[14];
  const float* dwc_w1 = (const float*)d_in[15];
  const float* dwc_b1 = (const float*)d_in[16];
  const float* dwc_w2 = (const float*)d_in[17];
  const float* dwc_b2 = (const float*)d_in[18];
  const float* aggr_w = (const float*)d_in[19];
  const float* aggr_b = (const float*)d_in[20];
  const float* ln2_g  = (const float*)d_in[21];
  const float* ln2_b  = (const float*)d_in[22];
  const float* fc1_w  = (const float*)d_in[23];
  const float* fc1_b  = (const float*)d_in[24];
  const float* dw_w   = (const float*)d_in[25];
  const float* dw_b   = (const float*)d_in[26];
  const float* fc2_w  = (const float*)d_in[27];
  const float* fc2_b  = (const float*)d_in[28];

  char* base = (char*)d_ws;
  float*     xn    = (float*)(base + OFF_XN);
  float*     fwR   = (float*)(base + OFF_FWR);
  float*     fwI   = (float*)(base + OFF_FWI);
  float*     f2R   = (float*)(base + OFF_F2R);
  float*     f2I   = (float*)(base + OFF_F2I);
  float*     band  = (float*)(base + OFF_BAND);   // reused as xm
  float*     xfft  = (float*)(base + OFF_XFFT);
  _Float16*  a16   = (_Float16*)(base + OFF_A16);
  _Float16*  w16   = (_Float16*)(base + OFF_W16);
  float*     sml   = (float*)(base + OFF_SML);
  _Float16*  t16   = (_Float16*)(base + OFF_T16); // overlays freed xn/fft scratch
  float*     y1    = f2R;                         // CAB mid (after FFT done)
  float*     convx = fwR;                         // CAB out (after FFT done)
  float*     lk    = f2I;                         // attn partial-conv out
  float*     xm    = band;

  const int FWCNT  = BB * HH * WF * DIM;          // 6,773,760
  const int IMGCNT = MROWS * DIM;                 // 13,271,040
  const int PADCNT = MROWS * KP180;               // 14,155,776
  const int GX     = MROWS / 256;                 // 288 blocks (8 waves x 32 rows)

  // 0) tables + weight conversion (deterministic each call)
  tables_k<<<55, 256, 0, stream>>>(sml);
  for (int k = 0; k < 3; ++k)
    wconv_k<<<(192*KP180+255)/256, 256, 0, stream>>>(band_w + (size_t)k*DIM*DIM,
        w16 + W16_BAND + (size_t)k*192*KP180, DIM, DIM, KP180, 192*KP180);
  wconv_k<<<(192*KP180+255)/256, 256, 0, stream>>>(aggr_w, w16 + W16_AGGR, DIM, DIM, KP180, 192*KP180);
  wconv_k<<<(FC1_NPAD*KP180+255)/256, 256, 0, stream>>>(fc1_w, w16 + W16_FC1, HID2, DIM, KP180, FC1_NPAD*KP180);
  wconv_k<<<(192*KP360+255)/256, 256, 0, stream>>>(fc2_w,  w16 + W16_FC2, DIM, HID, KP360, 192*KP360);

  // 1) LN1 -> xn (channel-last image) and x_fft accumulator init
  ln1_k<<<MROWS/8, 256, 0, stream>>>(x, ln1_g, ln1_b, xn, xfft);

  // 2) gates
  meanpix_k<<<BB*DIM, 256, 0, stream>>>(xn, sml + SM_IMGM, DIM, DIM);
  gates_k<<<1, 32, 0, stream>>>(sml, gate_w, gate_b, sml);

  // 3) forward 2D DFT (separable matmuls)
  dftw_k<<<FWCNT/256, 256, 0, stream>>>(xn, sml, fwR, fwI);
  dfth_k<<<FWCNT/256, 256, 0, stream>>>(fwR, fwI, sml, f2R, f2I);

  // 4) per band: masked inverse, then WMMA band-mixing GEMM accumulated into xfft
  for (int k = 0; k < 3; ++k) {
    invh_k<<<FWCNT/256, 256, 0, stream>>>(f2R, f2I, sml, fwR, fwI, k);
    invw_k<<<IMGCNT/256, 256, 0, stream>>>(fwR, fwI, sml, band);
    cvt16_k<<<PADCNT/256, 256, 0, stream>>>(band, a16);
    gemm_wmma_k<<<dim3(GX, 6), 256, 0, stream>>>(
        a16, w16 + W16_BAND + (size_t)k*192*KP180, KP180, DIM, /*mode*/0,
        band_b + (size_t)k*DIM, nullptr, nullptr, xfft, nullptr,
        sml + SM_GATE, k, 0);
  }

  // 5) CAB branch on x_fft
  cab1_k<<<(MROWS*CR)/256, 256, 0, stream>>>(xfft, cab_w1, cab_b1, y1);
  cab2_k<<<IMGCNT/256, 256, 0, stream>>>(y1, cab_w2, cab_b2, convx);
  meanpix_k<<<BB*DIM, 256, 0, stream>>>(convx, sml + SM_GAP, DIM, DIM);
  eca_k<<<1, 256, 0, stream>>>(sml, eca_w, sml);
  ecascale_k<<<IMGCNT/256, 256, 0, stream>>>(convx, sml);

  // 6) attention branch -> aggr GEMM fused with residual -> xm
  meanpix_k<<<BB*PDIM, 256, 0, stream>>>(xn, sml + SM_X1M, PDIM, DIM);
  dwcmlp_k<<<1, 256, 0, stream>>>(sml, dwc_w1, dwc_b1, dwc_w2, dwc_b2, sml);
  lkdyn_k<<<(MROWS*PDIM)/256, 256, 0, stream>>>(xn, plk, sml, lk);
  ya16_k<<<PADCNT/256, 256, 0, stream>>>(lk, xn, a16);
  gemm_wmma_k<<<dim3(GX, 6), 256, 0, stream>>>(
      a16, w16 + W16_AGGR, KP180, DIM, /*mode*/1,
      aggr_b, x, convx, xm, nullptr, nullptr, 0, 0);

  // 7) FFN: LN2 -> fc1 WMMA -> depthwise GLU -> fc2 WMMA (+final residual)
  ln2_k<<<MROWS/8, 256, 0, stream>>>(xm, ln2_g, ln2_b, a16);
  gemm_wmma_k<<<dim3(GX, 23), 256, 0, stream>>>(
      a16, w16 + W16_FC1, KP180, HID2, /*mode*/2,
      fc1_b, nullptr, nullptr, nullptr, t16, nullptr, 0, HID2);
  dwglu_k<<<(MROWS*KP360)/256, 256, 0, stream>>>(t16, dw_w, dw_b, a16);
  gemm_wmma_k<<<dim3(GX, 6), 256, 0, stream>>>(
      a16, w16 + W16_FC2, KP360, DIM, /*mode*/3,
      fc2_b, xm, nullptr, (float*)d_out, nullptr, nullptr, 0, 0);
}
